// MoELayer_29197187678928
// MI455X (gfx1250) — compile-verified, matrix operand order checked by reading, not run on hardware
//
#include <hip/hip_runtime.h>
#include <hip/hip_bf16.h>

// ---------------- problem constants ----------------
#define NEXP   8
#define TOPK   2
#define DDIM   1024
#define HDIM   1024
#define NTOK   2048          // B*T
#define SLOTS  (NTOK*TOPK)   // 4096 compact routed slots

typedef __attribute__((ext_vector_type(16))) __bf16 v16bf;
typedef __attribute__((ext_vector_type(8)))  float  v8f;

// ---------------- fp32 -> bf16 convert ----------------
__global__ void cvt_bf16(const float* __restrict__ in, __bf16* __restrict__ out, int n4) {
    int i = blockIdx.x * blockDim.x + threadIdx.x;
    if (i >= n4) return;
    float4 f = ((const float4*)in)[i];
    union { __bf16 b[4]; uint2 u; } t;
    t.b[0] = (__bf16)f.x; t.b[1] = (__bf16)f.y; t.b[2] = (__bf16)f.z; t.b[3] = (__bf16)f.w;
    ((uint2*)out)[i] = t.u;
}

// ---------------- router: logits, top-2, sigmoid gates ----------------
__global__ void router_kernel(const float* __restrict__ x, const float* __restrict__ gw,
                              const float* __restrict__ lb, int* __restrict__ idx,
                              float* __restrict__ wgt) {
    int wave = threadIdx.x >> 5;
    int lane = threadIdx.x & 31;
    int n = blockIdx.x * 8 + wave;
    if (n >= NTOK) return;
    const float* xr = x + (long)n * DDIM;
    float lg[NEXP];
#pragma unroll
    for (int e = 0; e < NEXP; ++e) {
        float s = 0.f;
        const float* gr = gw + (long)e * DDIM;
        for (int d = lane; d < DDIM; d += 32) s += xr[d] * gr[d];
#pragma unroll
        for (int off = 16; off > 0; off >>= 1) s += __shfl_xor(s, off, 32);
        lg[e] = s;
    }
    if (lane == 0) {
        float best = -1e30f; int bi = 0;
#pragma unroll
        for (int e = 0; e < NEXP; ++e) { float v = lg[e] + lb[e]; if (v > best) { best = v; bi = e; } }
        float best2 = -1e30f; int b2 = 0;
#pragma unroll
        for (int e = 0; e < NEXP; ++e) {
            if (e == bi) continue;
            float v = lg[e] + lb[e]; if (v > best2) { best2 = v; b2 = e; }
        }
        float s0 = 1.f / (1.f + __expf(-lg[bi]));   // gate uses raw logits
        float s1 = 1.f / (1.f + __expf(-lg[b2]));
        float inv = 1.f / (s0 + s1 + 1e-6f);
        idx[n * 2 + 0] = bi; idx[n * 2 + 1] = b2;
        wgt[n * 2 + 0] = s0 * inv; wgt[n * 2 + 1] = s1 * inv;
    }
}

// ---------------- deterministic compact expert token lists ----------------
__global__ void build_lists(const int* __restrict__ idx, int* __restrict__ tlist,
                            int* __restrict__ slot_of, int* __restrict__ counts,
                            int* __restrict__ bases) {
    __shared__ int sd[256];
    __shared__ int sbase;
    int tid = threadIdx.x;
    if (tid == 0) sbase = 0;
    __syncthreads();
    for (int e = 0; e < NEXP; ++e) {
        if (tid == 0) bases[e] = sbase;
        __syncthreads();
        for (int c = 0; c < NTOK; c += 256) {
            int n = c + tid;
            int k = -1;
            if (idx[2 * n] == e) k = 0;
            else if (idx[2 * n + 1] == e) k = 1;
            int f = (k >= 0) ? 1 : 0;
            sd[tid] = f;
            __syncthreads();
            for (int off = 1; off < 256; off <<= 1) {
                int v = sd[tid];
                int a = (tid >= off) ? sd[tid - off] : 0;
                __syncthreads();
                sd[tid] = v + a;
                __syncthreads();
            }
            int incl = sd[tid];
            int tot = sd[255];
            if (k >= 0) {
                int pos = sbase + incl - 1;
                tlist[pos] = n;
                slot_of[2 * n + k] = pos;
            }
            __syncthreads();
            if (tid == 0) sbase += tot;
            __syncthreads();
        }
        if (tid == 0) counts[e] = sbase - bases[e];
        __syncthreads();
    }
}

// ---------------- WMMA bf16 GEMM: C = act(A * B^T), double-buffered LDS ----------------
// A: rows (optionally gathered) x Kdim bf16, row stride == Kdim
// B: Ndim x Kdim bf16 (row-major), so B^T columns are contiguous rows
// RELU2: out = bf16(relu(c)^2) into Obf; else out = f32 c into Of.
#define TLM 128
#define TLN 128
#define KC  32
#define LDAH 40   // halves per LDS row (32 data + 8 pad) -> 80B, 16B aligned

__device__ inline v16bf mk16(uint4 lo, uint4 hi) {
    union { uint4 u[2]; v16bf v; } t;
    t.u[0] = lo; t.u[1] = hi;
    return t.v;
}

template <bool RELU2, bool GATHER>
__global__ void gemm_bf16(const __bf16* __restrict__ A, const __bf16* __restrict__ Bw,
                          __bf16* __restrict__ Obf, float* __restrict__ Of,
                          const int* __restrict__ tlist, const int* __restrict__ counts,
                          const int* __restrict__ bases,
                          int Mfixed, int Kdim, int Ndim, long expertBStride) {
    // two ping-pong buffers, each: A tile (128 x 32) + B tile (128 x 32), padded rows
    __shared__ __bf16 lds[2][(TLM + TLN) * LDAH];

    int e = blockIdx.z;
    int count = counts ? counts[e] : Mfixed;
    int base  = bases  ? bases[e]  : 0;
    int m0 = blockIdx.y * TLM;
    if (m0 >= count) return;
    int n0 = blockIdx.x * TLN;
    const __bf16* Bp = Bw + (long)e * expertBStride;

    int tid  = threadIdx.x;
    int wave = tid >> 5;
    int lane = tid & 31;
    int wm = wave & 1;          // 0..1 -> 64-row strip
    int wn = wave >> 1;         // 0..3 -> 32-col strip
    int lr = lane & 15;
    int hi = lane >> 4;
    int koff = hi * 8;          // halves

    v8f acc[4][2] = {};

    // ---- loop-invariant tile-row setup (no divergence in the K loop) ----
    int r    = tid & 127;
    bool isB = tid >= 128;
    const __bf16* srcRow;
    if (!isB) {
        int mrow = m0 + r;
        if (mrow >= count) mrow = count - 1;   // clamp: dup rows feed dead C rows only
        long rowIdx = GATHER ? (long)tlist[base + mrow] : (long)(base + mrow);
        srcRow = A + rowIdx * (long)Kdim;
    } else {
        srcRow = Bp + (long)(n0 + r) * Kdim;
    }
    const uint4* gp = (const uint4*)srcRow;        // 4 x uint4 per K-step
    int ldsRowOff = (isB ? TLM * LDAH : 0) + r * LDAH;

    uint4 d0, d1, d2, d3;

    auto stTile = [&](__bf16* buf) {
        uint4* q = (uint4*)(buf + ldsRowOff);
        q[0] = d0; q[1] = d1; q[2] = d2; q[3] = d3;
    };
    auto compute = [&](const __bf16* bufA) {
        const __bf16* bufB = bufA + TLM * LDAH;
        v16bf af[4], bfr[2];
#pragma unroll
        for (int mi = 0; mi < 4; ++mi) {
            const __bf16* pa = bufA + (wm * 64 + mi * 16 + lr) * LDAH + koff;
            af[mi] = mk16(*(const uint4*)pa, *(const uint4*)(pa + 16));
        }
#pragma unroll
        for (int ni = 0; ni < 2; ++ni) {
            const __bf16* pb = bufB + (wn * 32 + ni * 16 + lr) * LDAH + koff;
            bfr[ni] = mk16(*(const uint4*)pb, *(const uint4*)(pb + 16));
        }
#pragma unroll
        for (int mi = 0; mi < 4; ++mi)
#pragma unroll
            for (int ni = 0; ni < 2; ++ni)
                acc[mi][ni] = __builtin_amdgcn_wmma_f32_16x16x32_bf16(
                    false, af[mi], false, bfr[ni], (short)0, acc[mi][ni], false, false);
    };

    // ---- software-pipelined K loop: load(it) overlaps compute(it-1) ----
    d0 = gp[0]; d1 = gp[1]; d2 = gp[2]; d3 = gp[3]; gp += 4;
    stTile(lds[0]);
    int nk = Kdim / KC;
    for (int it = 1; it < nk; ++it) {
        __syncthreads();
        uint4 t0 = gp[0], t1 = gp[1], t2 = gp[2], t3 = gp[3]; gp += 4;  // in flight
        compute((it & 1) ? lds[0] : lds[1]);                            // buf[(it-1)&1]
        d0 = t0; d1 = t1; d2 = t2; d3 = t3;
        stTile((it & 1) ? lds[1] : lds[0]);                             // buf[it&1]
    }
    __syncthreads();
    compute((nk & 1) ? lds[0] : lds[1]);                                // buf[(nk-1)&1]

    // ---- epilogue: C element (vgpr v, lane L) -> row = v + (L>=16?8:0), col = L%16 ----
#pragma unroll
    for (int mi = 0; mi < 4; ++mi) {
#pragma unroll
        for (int ni = 0; ni < 2; ++ni) {
            int rowl = wm * 64 + mi * 16 + hi * 8;
            int col  = n0 + wn * 32 + ni * 16 + lr;
#pragma unroll
            for (int v = 0; v < 8; ++v) {
                int mrow = m0 + rowl + v;
                if (mrow < count) {
                    float c = acc[mi][ni][v];
                    long o = (long)(base + mrow) * Ndim + col;
                    if (RELU2) {
                        float rr = fmaxf(c, 0.f);
                        Obf[o] = (__bf16)(rr * rr);
                    } else {
                        Of[o] = c;
                    }
                }
            }
        }
    }
}

// ---------------- combine: out += sum_k w[n,k] * expert_out[slot(n,k)] ----------------
__global__ void combine_kernel(float* __restrict__ out, const float* __restrict__ eo,
                               const int* __restrict__ slot_of, const float* __restrict__ wgt) {
    int t = blockIdx.x * blockDim.x + threadIdx.x;   // NTOK*DDIM/4 threads
    int n  = t / (DDIM / 4);
    int dc = (t % (DDIM / 4)) * 4;
    float4 o = *(const float4*)(out + (long)n * DDIM + dc);
#pragma unroll
    for (int k = 0; k < TOPK; ++k) {
        int s  = slot_of[n * 2 + k];
        float w = wgt[n * 2 + k];
        float4 e4 = *(const float4*)(eo + (long)s * DDIM + dc);
        o.x += w * e4.x; o.y += w * e4.y; o.z += w * e4.z; o.w += w * e4.w;
    }
    *(float4*)(out + (long)n * DDIM + dc) = o;
}

// ---------------- launch ----------------
extern "C" void kernel_launch(void* const* d_in, const int* in_sizes, int n_in,
                              void* d_out, int out_size, void* d_ws, size_t ws_size,
                              hipStream_t stream) {
    const float* x      = (const float*)d_in[0];
    const float* gate_w = (const float*)d_in[1];
    const float* lbbias = (const float*)d_in[2];
    const float* w1     = (const float*)d_in[3];
    const float* w2     = (const float*)d_in[4];
    const float* cfc    = (const float*)d_in[5];
    const float* cproj  = (const float*)d_in[6];
    float* out = (float*)d_out;

    char* w = (char*)d_ws;
    size_t off = 0;
    auto alloc = [&](size_t bytes) -> void* {
        off = (off + 255) & ~(size_t)255;
        void* p = w + off;
        off += bytes;
        return p;
    };
    __bf16* xb     = (__bf16*)alloc((size_t)NTOK * DDIM * 2);
    __bf16* w1b    = (__bf16*)alloc((size_t)NEXP * HDIM * DDIM * 2);
    __bf16* w2b    = (__bf16*)alloc((size_t)NEXP * DDIM * HDIM * 2);
    __bf16* cfcb   = (__bf16*)alloc((size_t)HDIM * DDIM * 2);
    __bf16* cprojb = (__bf16*)alloc((size_t)DDIM * HDIM * 2);
    int*    idx    = (int*)  alloc((size_t)NTOK * TOPK * 4);
    float*  wgt    = (float*)alloc((size_t)NTOK * TOPK * 4);
    int*    tlist  = (int*)  alloc((size_t)SLOTS * 4);
    int*    slotof = (int*)  alloc((size_t)NTOK * TOPK * 4);
    int*    counts = (int*)  alloc((size_t)NEXP * 4);
    int*    bases  = (int*)  alloc((size_t)NEXP * 4);
    __bf16* hid_e  = (__bf16*)alloc((size_t)SLOTS * HDIM * 2);
    __bf16* hid_sh = (__bf16*)alloc((size_t)NTOK * HDIM * 2);
    float*  eo     = (float*)alloc((size_t)SLOTS * DDIM * 4);

    // 1. convert to bf16
    {
        int n4;
        n4 = NTOK * DDIM / 4;        cvt_bf16<<<(n4 + 255) / 256, 256, 0, stream>>>(x, xb, n4);
        n4 = NEXP * HDIM * DDIM / 4; cvt_bf16<<<(n4 + 255) / 256, 256, 0, stream>>>(w1, w1b, n4);
        n4 = NEXP * DDIM * HDIM / 4; cvt_bf16<<<(n4 + 255) / 256, 256, 0, stream>>>(w2, w2b, n4);
        n4 = HDIM * DDIM / 4;        cvt_bf16<<<(n4 + 255) / 256, 256, 0, stream>>>(cfc, cfcb, n4);
        n4 = DDIM * HDIM / 4;        cvt_bf16<<<(n4 + 255) / 256, 256, 0, stream>>>(cproj, cprojb, n4);
    }
    // 2. router
    router_kernel<<<NTOK / 8, 256, 0, stream>>>(x, gate_w, lbbias, idx, wgt);
    // 3. deterministic expert lists
    build_lists<<<1, 256, 0, stream>>>(idx, tlist, slotof, counts, bases);
    // 4. shared expert GEMM1: hid_sh = relu(x @ cfc^T)^2
    gemm_bf16<true, false><<<dim3(HDIM / TLN, NTOK / TLM, 1), 256, 0, stream>>>(
        xb, cfcb, hid_sh, nullptr, nullptr, nullptr, nullptr, NTOK, DDIM, HDIM, 0);
    // 5. routed experts GEMM1 (gathered): hid_e[slot] = relu(x[tok] @ w1[e]^T)^2
    gemm_bf16<true, true><<<dim3(HDIM / TLN, NTOK / TLM, NEXP), 256, 0, stream>>>(
        xb, w1b, hid_e, nullptr, tlist, counts, bases, 0, DDIM, HDIM,
        (long)HDIM * DDIM);
    // 6. shared expert GEMM2 -> writes full d_out (base of the sum)
    gemm_bf16<false, false><<<dim3(DDIM / TLN, NTOK / TLM, 1), 256, 0, stream>>>(
        hid_sh, cprojb, nullptr, out, nullptr, nullptr, nullptr, NTOK, HDIM, DDIM, 0);
    // 7. routed experts GEMM2: eo[slot] = hid_e[slot] @ w2[e]^T
    gemm_bf16<false, false><<<dim3(DDIM / TLN, NTOK / TLM, NEXP), 256, 0, stream>>>(
        hid_e, w2b, nullptr, eo, nullptr, counts, bases, 0, HDIM, DDIM,
        (long)DDIM * HDIM);
    // 8. combine with gate weights (deterministic, no atomics)
    combine_kernel<<<(NTOK * DDIM / 4) / 256, 256, 0, stream>>>(out, eo, slotof, wgt);
}
